// ImprovedEdgeGNN_64209761075597
// MI455X (gfx1250) — compile-verified, matrix operand order checked by reading, not run on hardware
//
#include <hip/hip_runtime.h>
#include <hip/hip_bf16.h>

typedef __attribute__((ext_vector_type(16))) _Float16 v16h;
typedef __attribute__((ext_vector_type(8)))  _Float16 v8h;
typedef __attribute__((ext_vector_type(4)))  _Float16 v4h;
typedef __attribute__((ext_vector_type(8)))  float    v8f;
typedef __attribute__((ext_vector_type(4)))  float    v4f;

#define CAT16(lo, hi) __builtin_shufflevector((lo), (hi), 0,1,2,3,4,5,6,7,8,9,10,11,12,13,14,15)

// ---------- helpers ----------
__device__ inline v8h cvt8(const float* p) {
    v4f a = *(const v4f*)p;
    v4f b = *(const v4f*)(p + 4);
    v8h r;
    r[0] = (_Float16)a[0]; r[1] = (_Float16)a[1]; r[2] = (_Float16)a[2]; r[3] = (_Float16)a[3];
    r[4] = (_Float16)b[0]; r[5] = (_Float16)b[1]; r[6] = (_Float16)b[2]; r[7] = (_Float16)b[3];
    return r;
}

__device__ inline v8h cvt8s(const float* p, float sc) {
    v4f a = *(const v4f*)p;
    v4f b = *(const v4f*)(p + 4);
    v8h r;
    r[0] = (_Float16)(a[0]*sc); r[1] = (_Float16)(a[1]*sc); r[2] = (_Float16)(a[2]*sc); r[3] = (_Float16)(a[3]*sc);
    r[4] = (_Float16)(b[0]*sc); r[5] = (_Float16)(b[1]*sc); r[6] = (_Float16)(b[2]*sc); r[7] = (_Float16)(b[3]*sc);
    return r;
}

// ---------- trivial kernels ----------
__global__ void k_zero(float* __restrict__ p, long long n) {
    long long i = (long long)blockIdx.x * blockDim.x + threadIdx.x;
    if (i < n) p[i] = 0.0f;
}

__global__ void k_count(const int* __restrict__ dst, float* __restrict__ cnt, int E) {
    int e = blockIdx.x * blockDim.x + threadIdx.x;
    if (e < E) atomicAdd(&cnt[dst[e]], 1.0f);
}

// scatter: s[dst[e], :] += x[src[e], :]   (4 channels per thread)
__global__ void k_scatter(const float* __restrict__ x, const int* __restrict__ src,
                          const int* __restrict__ dst, float* __restrict__ s,
                          int C4, int C, int E) {
    long long i = (long long)blockIdx.x * blockDim.x + threadIdx.x;
    long long tot = (long long)E * C4;
    if (i >= tot) return;
    int e = (int)(i / C4);
    int q = (int)(i % C4);
    int sr = src[e], dr = dst[e];
    v4f v = *(const v4f*)(x + (long long)sr * C + q * 4);
    float* o = s + (long long)dr * C + q * 4;
    atomicAdd(o + 0, v[0]);
    atomicAdd(o + 1, v[1]);
    atomicAdd(o + 2, v[2]);
    atomicAdd(o + 3, v[3]);
}

// ---------- SAGE layer: pre = [agg | x] @ [Wl|Wr]^T + bl ; + BN batch stats ----------
__global__ __launch_bounds__(256) void k_sage(
        const float* __restrict__ s, const float* __restrict__ x, const float* __restrict__ cnt,
        const float* __restrict__ Wl, const float* __restrict__ Wr, const float* __restrict__ bl,
        float* __restrict__ pre, float* __restrict__ stS, float* __restrict__ stQ,
        int N, int Cin) {
    const int K = 2 * Cin;                  // 256 (layer1) or 128 (layer2)
    __shared__ _Float16 Wlds[64 * 256];     // [64][K] f16, row-major stride K
    for (int i = threadIdx.x; i < 64 * K; i += blockDim.x) {
        int o = i / K, k = i % K;
        float w = (k < Cin) ? Wl[o * Cin + k] : Wr[o * Cin + (k - Cin)];
        Wlds[i] = (_Float16)w;
    }
    __syncthreads();

    int wid = threadIdx.x >> 5, lane = threadIdx.x & 31;
    int r = lane & 15, hi = lane >> 4;
    int nT = (N + 15) >> 4;
    int tile = blockIdx.x * (blockDim.x >> 5) + wid;
    if (tile >= nT) return;

    int row = tile * 16 + r;
    if (row >= N) row = N - 1;
    float dinv = 1.0f / fmaxf(cnt[row], 1.0f);

    const float* srow = s + (long long)row * Cin;
    const float* xrow = x + (long long)row * Cin;

    float bias[4];
#pragma unroll
    for (int nt = 0; nt < 4; nt++) bias[nt] = bl[nt * 16 + r];

    v8f acc[4] = {};
    const int ks1 = Cin >> 5;               // k-steps per region (uniform, branch-free)
    // region 1: neighbor aggregate (scaled by 1/deg)
    for (int kk = 0; kk < ks1; kk++) {
        int f0 = kk * 32 + hi * 8;
        v16h A = CAT16(cvt8s(srow + f0, dinv), cvt8s(srow + f0 + 16, dinv));
#pragma unroll
        for (int nt = 0; nt < 4; nt++) {
            const _Float16* wr = Wlds + (nt * 16 + r) * K + f0;
            v16h B = CAT16(*(const v8h*)(wr), *(const v8h*)(wr + 16));
            acc[nt] = __builtin_amdgcn_wmma_f32_16x16x32_f16(
                false, A, false, B, (short)0, acc[nt], false, false);
        }
    }
    // region 2: self features
    for (int kk = 0; kk < ks1; kk++) {
        int f0 = kk * 32 + hi * 8;
        v16h A = CAT16(cvt8(xrow + f0), cvt8(xrow + f0 + 16));
#pragma unroll
        for (int nt = 0; nt < 4; nt++) {
            const _Float16* wr = Wlds + (nt * 16 + r) * K + Cin + f0;
            v16h B = CAT16(*(const v8h*)(wr), *(const v8h*)(wr + 16));
            acc[nt] = __builtin_amdgcn_wmma_f32_16x16x32_f16(
                false, A, false, B, (short)0, acc[nt], false, false);
        }
    }

    // epilogue: store pre-BN activations + local column sums/sumsq
    float ls[4], lq[4];
#pragma unroll
    for (int nt = 0; nt < 4; nt++) {
        ls[nt] = 0.0f; lq[nt] = 0.0f;
#pragma unroll
        for (int j = 0; j < 8; j++) {
            int m = j + hi * 8;
            int rr = tile * 16 + m;
            float v = acc[nt][j] + bias[nt];
            if (rr < N) {
                pre[(long long)rr * 64 + nt * 16 + r] = v;
                ls[nt] += v;
                lq[nt] += v * v;
            }
        }
    }
#pragma unroll
    for (int nt = 0; nt < 4; nt++) {
        float sv = ls[nt] + __shfl_down(ls[nt], 16);
        float qv = lq[nt] + __shfl_down(lq[nt], 16);
        if (hi == 0) {
            atomicAdd(&stS[nt * 16 + r], sv);
            atomicAdd(&stQ[nt * 16 + r], qv);
        }
    }
}

__global__ void k_bnfin(const float* __restrict__ stS, const float* __restrict__ stQ,
                        const float* __restrict__ g, const float* __restrict__ b,
                        float* __restrict__ ss, int N) {
    int c = threadIdx.x;
    if (c >= 64) return;
    float m = stS[c] / (float)N;
    float v = stQ[c] / (float)N - m * m;
    float sc = g[c] * rsqrtf(v + 1e-5f);
    ss[c] = sc;
    ss[64 + c] = b[c] - m * sc;
}

// BN+ReLU apply, vectorized by 4 (rows are 64 channels -> groups of 4 stay in-row)
__global__ void k_bnrelu_f32(const float* __restrict__ pre, const float* __restrict__ ss,
                             float* __restrict__ h, long long n4) {
    long long i = (long long)blockIdx.x * blockDim.x + threadIdx.x;
    if (i >= n4) return;
    long long idx = i * 4;
    int c = (int)(idx & 63);
    v4f p  = *(const v4f*)(pre + idx);
    v4f sc = *(const v4f*)(ss + c);
    v4f sh = *(const v4f*)(ss + 64 + c);
    v4f r;
#pragma unroll
    for (int j = 0; j < 4; j++) r[j] = fmaxf(p[j] * sc[j] + sh[j], 0.0f);
    *(v4f*)(h + idx) = r;
}

__global__ void k_bnrelu_f16(const float* __restrict__ pre, const float* __restrict__ ss,
                             _Float16* __restrict__ h, long long n4) {
    long long i = (long long)blockIdx.x * blockDim.x + threadIdx.x;
    if (i >= n4) return;
    long long idx = i * 4;
    int c = (int)(idx & 63);
    v4f p  = *(const v4f*)(pre + idx);
    v4f sc = *(const v4f*)(ss + c);
    v4f sh = *(const v4f*)(ss + 64 + c);
    v4h r;
#pragma unroll
    for (int j = 0; j < 4; j++) r[j] = (_Float16)fmaxf(p[j] * sc[j] + sh[j], 0.0f);
    *(v4h*)(h + idx) = r;
}

// ---------- fused edge MLP: 3 layers per 16-edge tile ----------
// feature layout (K=152, padded to 160): [h2[src](64) | h2[dst](64) | edge_attr(16) | emb(8) | pad(8)]
__global__ __launch_bounds__(256) void k_edge(
        const _Float16* __restrict__ h2, const int* __restrict__ src, const int* __restrict__ dst,
        const float* __restrict__ ea, const int* __restrict__ tix, const float* __restrict__ temb,
        const float* __restrict__ Wm1, const float* __restrict__ bm1,
        const float* __restrict__ Wm2, const float* __restrict__ bm2,
        const float* __restrict__ Wm3, const float* __restrict__ bm3,
        float* __restrict__ out, int E) {
    __shared__ _Float16 W1lds[64 * 160];   // Wm1 as [64][160] f16, zero-padded K
    __shared__ _Float16 W2lds[32 * 64];    // Wm2 as [32][64] f16
    __shared__ float    W3lds[2 * 32];
    __shared__ float    b3lds[2];
    __shared__ _Float16 Z1[8][16 * 64];    // per-wave relayout tile (D -> A)
    __shared__ float    Z2[8][16 * 32];

    for (int i = threadIdx.x; i < 64 * 160; i += blockDim.x) {
        int o = i / 160, k = i % 160;
        W1lds[i] = (_Float16)((k < 152) ? Wm1[o * 152 + k] : 0.0f);
    }
    for (int i = threadIdx.x; i < 32 * 64; i += blockDim.x) W2lds[i] = (_Float16)Wm2[i];
    if (threadIdx.x < 64) W3lds[threadIdx.x] = Wm3[threadIdx.x];
    if (threadIdx.x < 2)  b3lds[threadIdx.x] = bm3[threadIdx.x];
    __syncthreads();

    int wid = threadIdx.x >> 5, lane = threadIdx.x & 31;
    int r = lane & 15, hi = lane >> 4;
    int el = lane >> 1, o = lane & 1;

    float bias1[4];
#pragma unroll
    for (int nt = 0; nt < 4; nt++) bias1[nt] = bm1[nt * 16 + r];
    float bias2[2];
#pragma unroll
    for (int nt = 0; nt < 2; nt++) bias2[nt] = bm2[nt * 16 + r];

    // loop-invariant register fragments: GEMM2 B, layer-3 weight row
    v16h B2[2][2];
#pragma unroll
    for (int kk = 0; kk < 2; kk++)
#pragma unroll
        for (int nt = 0; nt < 2; nt++) {
            const _Float16* wr = W2lds + (nt * 16 + r) * 64 + kk * 32 + hi * 8;
            B2[kk][nt] = CAT16(*(const v8h*)(wr), *(const v8h*)(wr + 16));
        }
    v4f w3r[8];
#pragma unroll
    for (int q = 0; q < 8; q++) w3r[q] = *(const v4f*)(W3lds + o * 32 + q * 4);

    int nT = (E + 15) >> 4;
    int tstride = gridDim.x * 8;
    for (int t = blockIdx.x * 8 + wid; t < nT; t += tstride) {
        int e = t * 16 + r;
        long long eg = (e < E) ? e : (E - 1);
        int sr = src[eg], dr = dst[eg], ti = tix[eg];

        // prefetch next tile's edge_attr stream (only HBM-resident operand)
        {
            long long en = (long long)(t + tstride) * 16 + r;
            if (en < E) __builtin_prefetch(ea + en * 16, 0, 0);
        }

        const _Float16* hsrc = h2 + (long long)sr * 64;
        const _Float16* hdst = h2 + (long long)dr * 64;

        // ---- GEMM1: [16 x 160] @ [160 x 64] -> [16 x 64]
        v8f acc[4] = {};
#pragma unroll
        for (int kk = 0; kk < 4; kk++) {                 // k-steps 0..3: h[src] | h[dst]
            const _Float16* base = (kk < 2) ? hsrc : hdst;   // compile-time after unroll
            int f0 = (kk & 1) * 32 + hi * 8;
            v16h A = CAT16(*(const v8h*)(base + f0), *(const v8h*)(base + f0 + 16));
            int fw = kk * 32 + hi * 8;
#pragma unroll
            for (int nt = 0; nt < 4; nt++) {
                const _Float16* wr = W1lds + (nt * 16 + r) * 160 + fw;
                v16h B = CAT16(*(const v8h*)(wr), *(const v8h*)(wr + 16));
                acc[nt] = __builtin_amdgcn_wmma_f32_16x16x32_f16(
                    false, A, false, B, (short)0, acc[nt], false, false);
            }
        }
        {                                                // k-step 4: edge_attr | emb | pad
            v8h a0 = cvt8(ea + eg * 16 + hi * 8);
            v8h embv = cvt8(temb + (long long)ti * 8);
            v8h zz = {};
            v8h a1 = hi ? zz : embv;                     // branchless lane select
            v16h A = CAT16(a0, a1);
            int fw = 128 + hi * 8;
#pragma unroll
            for (int nt = 0; nt < 4; nt++) {
                const _Float16* wr = W1lds + (nt * 16 + r) * 160 + fw;
                v16h B = CAT16(*(const v8h*)(wr), *(const v8h*)(wr + 16));
                acc[nt] = __builtin_amdgcn_wmma_f32_16x16x32_f16(
                    false, A, false, B, (short)0, acc[nt], false, false);
            }
        }

        // relu + bias, relayout D -> A via per-wave LDS
        _Float16* z1 = Z1[wid];
#pragma unroll
        for (int nt = 0; nt < 4; nt++)
#pragma unroll
            for (int j = 0; j < 8; j++) {
                int m = j + hi * 8;
                z1[m * 64 + nt * 16 + r] = (_Float16)fmaxf(acc[nt][j] + bias1[nt], 0.0f);
            }
        __builtin_amdgcn_wave_barrier();

        // ---- GEMM2: [16 x 64] @ [64 x 32] -> [16 x 32]
        v8f acc2[2] = {};
#pragma unroll
        for (int kk = 0; kk < 2; kk++) {
            int f0 = kk * 32 + hi * 8;
            v16h A2 = CAT16(*(const v8h*)(z1 + r * 64 + f0), *(const v8h*)(z1 + r * 64 + f0 + 16));
#pragma unroll
            for (int nt = 0; nt < 2; nt++) {
                acc2[nt] = __builtin_amdgcn_wmma_f32_16x16x32_f16(
                    false, A2, false, B2[kk][nt], (short)0, acc2[nt], false, false);
            }
        }

        float* z2 = Z2[wid];
#pragma unroll
        for (int nt = 0; nt < 2; nt++)
#pragma unroll
            for (int j = 0; j < 8; j++) {
                int m = j + hi * 8;
                z2[m * 32 + nt * 16 + r] = fmaxf(acc2[nt][j] + bias2[nt], 0.0f);
            }
        __builtin_amdgcn_wave_barrier();

        // ---- layer 3: [16 x 32] @ [32 x 2] via VALU dot (one (edge,out) pair per lane)
        float sum = b3lds[o];
#pragma unroll
        for (int q = 0; q < 8; q++) {
            v4f zv = *(const v4f*)(z2 + el * 32 + q * 4);
            sum += zv[0] * w3r[q][0] + zv[1] * w3r[q][1] + zv[2] * w3r[q][2] + zv[3] * w3r[q][3];
        }
        int eo = t * 16 + el;
        if (eo < E) out[(long long)eo * 2 + o] = sum;
    }
}

// ---------- launcher ----------
static inline int nblk(long long n, int b) { return (int)((n + b - 1) / b); }

extern "C" void kernel_launch(void* const* d_in, const int* in_sizes, int n_in,
                              void* d_out, int out_size, void* d_ws, size_t ws_size,
                              hipStream_t stream) {
    const float* x     = (const float*)d_in[0];
    const int*   ei    = (const int*)d_in[1];
    const float* eattr = (const float*)d_in[2];
    const int*   tix   = (const int*)d_in[3];
    const float* temb  = (const float*)d_in[4];
    const float* W1l = (const float*)d_in[5];  const float* b1l = (const float*)d_in[6];
    const float* W1r = (const float*)d_in[7];
    const float* g1  = (const float*)d_in[8];  const float* be1 = (const float*)d_in[9];
    const float* W2l = (const float*)d_in[10]; const float* b2l = (const float*)d_in[11];
    const float* W2r = (const float*)d_in[12];
    const float* g2  = (const float*)d_in[13]; const float* be2 = (const float*)d_in[14];
    const float* Wm1 = (const float*)d_in[15]; const float* bm1 = (const float*)d_in[16];
    const float* Wm2 = (const float*)d_in[17]; const float* bm2 = (const float*)d_in[18];
    const float* Wm3 = (const float*)d_in[19]; const float* bm3 = (const float*)d_in[20];

    const int CIN = 128;
    const int N = in_sizes[0] / CIN;
    const int E = in_sizes[1] / 2;
    const int* src  = ei;
    const int* dstp = ei + E;

    // workspace layout
    char* ws = (char*)d_ws;
    float*     cnt  = (float*)ws;                                   // N
    float*     sbuf = (float*)(ws + (size_t)N * 4);                 // N*128 (reused N*64 for layer2)
    float*     pre  = sbuf + (size_t)N * 128;                       // N*64 (reused both layers)
    float*     h1   = pre + (size_t)N * 64;                         // N*64
    _Float16*  h2   = (_Float16*)(h1 + (size_t)N * 64);             // N*64 f16
    float*     st   = (float*)((char*)h2 + (size_t)N * 64 * 2);     // 512 floats
    float *stS1 = st,       *stQ1 = st + 64,  *ss1 = st + 128;
    float *stS2 = st + 256, *stQ2 = st + 320, *ss2 = st + 384;

    // ---- layer 1
    k_zero<<<nblk(N, 256), 256, 0, stream>>>(cnt, N);
    k_zero<<<nblk((long long)N * 128, 256), 256, 0, stream>>>(sbuf, (long long)N * 128);
    k_zero<<<2, 256, 0, stream>>>(st, 512);
    k_count<<<nblk(E, 256), 256, 0, stream>>>(dstp, cnt, E);
    k_scatter<<<nblk((long long)E * 32, 256), 256, 0, stream>>>(x, src, dstp, sbuf, 32, 128, E);
    int nT = (N + 15) / 16;
    k_sage<<<nblk(nT, 8), 256, 0, stream>>>(sbuf, x, cnt, W1l, W1r, b1l, pre, stS1, stQ1, N, 128);
    k_bnfin<<<1, 64, 0, stream>>>(stS1, stQ1, g1, be1, ss1, N);
    k_bnrelu_f32<<<nblk((long long)N * 16, 256), 256, 0, stream>>>(pre, ss1, h1, (long long)N * 16);

    // ---- layer 2
    k_zero<<<nblk((long long)N * 64, 256), 256, 0, stream>>>(sbuf, (long long)N * 64);
    k_scatter<<<nblk((long long)E * 16, 256), 256, 0, stream>>>(h1, src, dstp, sbuf, 16, 64, E);
    k_sage<<<nblk(nT, 8), 256, 0, stream>>>(sbuf, h1, cnt, W2l, W2r, b2l, pre, stS2, stQ2, N, 64);
    k_bnfin<<<1, 64, 0, stream>>>(stS2, stQ2, g2, be2, ss2, N);
    k_bnrelu_f16<<<nblk((long long)N * 16, 256), 256, 0, stream>>>(pre, ss2, h2, (long long)N * 16);

    // ---- fused edge MLP
    int nTe = (E + 15) / 16;
    int grid = (nTe + 7) / 8;
    if (grid > 4096) grid = 4096;
    k_edge<<<grid, 256, 0, stream>>>(h2, src, dstp, eattr, tix, temb,
                                     Wm1, bm1, Wm2, bm2, Wm3, bm3, (float*)d_out, E);
}
